// GNNEncoder_85461259256118
// MI455X (gfx1250) — matrix operand order repaired and checked by reading, not compiled, not run on hardware
//
#include <hip/hip_runtime.h>

#define D 128

typedef __attribute__((ext_vector_type(16))) __bf16 v16bf;
typedef __attribute__((ext_vector_type(8)))  float  v8f;
typedef __attribute__((ext_vector_type(16))) unsigned short v16u;
typedef __attribute__((ext_vector_type(8)))  unsigned short v8u;

// ---------- helpers ----------
__device__ __forceinline__ unsigned short f2bf(float f) {
    unsigned u = __float_as_uint(f);
    u += 0x7FFFu + ((u >> 16) & 1u);          // round to nearest even
    return (unsigned short)(u >> 16);
}
__device__ __forceinline__ unsigned fkey(float f) {   // order-preserving float->uint
    unsigned u = __float_as_uint(f);
    return (u & 0x80000000u) ? ~u : (u | 0x80000000u);
}
__device__ __forceinline__ float keyf(unsigned k) {
    unsigned u = (k & 0x80000000u) ? (k & 0x7FFFFFFFu) : ~k;
    return __uint_as_float(u);
}
__device__ __forceinline__ float lrelu(float x) { return x > 0.0f ? x : 0.2f * x; }

// ---------- prep: u_s[t] = Ws[t] @ a_s[t], u_d[t] = Wd[t] @ a_d[t] ----------
__global__ void prep_u(const float* __restrict__ Ws, const float* __restrict__ Wd,
                       const float* __restrict__ a_s, const float* __restrict__ a_d,
                       float* __restrict__ us, float* __restrict__ ud)
{
    int j = blockIdx.x * blockDim.x + threadIdx.x;   // 0..767
    if (j >= 6 * D) return;
    int d = j & (D - 1);
    int which = j >> 7;                              // 0..5
    int t = which >> 1;
    bool isd = which & 1;
    const float* W = isd ? Wd : Ws;
    const float* a = isd ? a_d : a_s;
    float* u = isd ? ud : us;
    float s = 0.f;
    for (int h = 0; h < D; ++h)
        s += W[(size_t)t * D * D + (size_t)d * D + h] * a[t * D + h];
    u[t * D + d] = s;
}

// ---------- prep: swizzle Ws into WMMA B-operand layout (bf16) ----------
// out flat index o = ((((t*4+kt)*8+nt)*32+lane)*16+j)
// lane: N = nt*16 + (lane&15); K = kt*32 + (lane>>4)*16 + j   (contiguous k order, same as A)
__global__ void swizzle_w(const float* __restrict__ Ws, unsigned short* __restrict__ wsw)
{
    int o = blockIdx.x * blockDim.x + threadIdx.x;
    if (o >= 3 * D * D) return;
    int j    = o & 15;
    int lane = (o >> 4) & 31;
    int nt   = (o >> 9) & 7;
    int kt   = (o >> 12) & 3;
    int t    = o >> 14;
    int k = kt * 32 + (lane >> 4) * 16 + j;
    int n = nt * 16 + (lane & 15);
    wsw[o] = f2bf(Ws[(size_t)t * D * D + (size_t)k * D + n]);
}

// ---------- prep: fp32 -> bf16 feature conversion ----------
__global__ void cvt_bf16(const float* __restrict__ x, unsigned short* __restrict__ xb, long n)
{
    long i = (long)blockIdx.x * blockDim.x + threadIdx.x;
    if (i < n) xb[i] = f2bf(x[i]);
}

// ---------- GEMM: H[M x 128] = bf16(X)[M x 128] @ bf16(W)[128 x 128] via WMMA ----------
// One wave owns a 16-row strip and all 8 N-tiles: 4 K-steps x 8 N-tiles = 32 v_wmma.
__global__ void gat_gemm_wmma(const unsigned short* __restrict__ Xb,
                              const unsigned short* __restrict__ Wsw,
                              float* __restrict__ H, int M)
{
    int gw   = (int)((blockIdx.x * blockDim.x + threadIdx.x) >> 5);
    int lane = (int)(threadIdx.x & 31);
    int m0 = gw * 16;
    if (m0 >= M) return;
    int half = lane >> 4;
    int mrow = m0 + (lane & 15);
    if (mrow >= M) mrow = M - 1;
    const unsigned short* xrow = Xb + (size_t)mrow * D;

    v8f c[8] = {};
#pragma unroll
    for (int kt = 0; kt < 4; ++kt) {
        int k0 = kt * 32;
        // A 16x32 bf16 layout: lanes 0-15 row m hold K = k0+{0..15}; lanes 16-31 hold {16..31}
        v8u lo = *(const v8u*)(xrow + k0 + half * 16);
        v8u hi = *(const v8u*)(xrow + k0 + half * 16 + 8);
        v16u au;
#pragma unroll
        for (int j = 0; j < 8; ++j) { au[j] = lo[j]; au[j + 8] = hi[j]; }
        v16bf av = __builtin_bit_cast(v16bf, au);
#pragma unroll
        for (int nt = 0; nt < 8; ++nt) {
            v16bf bv = *(const v16bf*)(Wsw + (size_t)(((kt * 8 + nt) * 32 + lane) * 16));
            c[nt] = __builtin_amdgcn_wmma_f32_16x16x32_bf16(
                false, av, false, bv, (short)0, c[nt], false, false);
        }
    }
    // C/D layout: lane holds N=(lane&15)+16*nt; VGPR r holds row m0 + r + 8*half
    int col0 = lane & 15;
#pragma unroll
    for (int nt = 0; nt < 8; ++nt) {
#pragma unroll
        for (int r = 0; r < 8; ++r) {
            int row = m0 + r + 8 * half;
            if (row < M) H[(size_t)row * D + nt * 16 + col0] = c[nt][r];
        }
    }
}

// ---------- alpha = X @ u   (one wave per row) ----------
__global__ void matvec(const float* __restrict__ X, const float* __restrict__ u,
                       float* __restrict__ outv, int M)
{
    int row  = (int)((blockIdx.x * blockDim.x + threadIdx.x) >> 5);
    int lane = (int)(threadIdx.x & 31);
    if (row >= M) return;
    float4 xv = ((const float4*)(X + (size_t)row * D))[lane];
    float4 uv = ((const float4*)u)[lane];
    float s = xv.x * uv.x + xv.y * uv.y + xv.z * uv.z + xv.w * uv.w;
#pragma unroll
    for (int o = 16; o > 0; o >>= 1) s += __shfl_xor(s, o);
    if (lane == 0) outv[row] = s;
}

// ---------- per-conv reset: acc=0 (nd*128), emax=-inf key, den=0 ----------
__global__ void reset_conv(unsigned* __restrict__ ek, float* __restrict__ den,
                           float* __restrict__ acc, int nd)
{
    long i = (long)blockIdx.x * blockDim.x + threadIdx.x;
    if (i >= (long)nd * D) return;
    acc[i] = 0.0f;
    if (i < nd) { ek[i] = 0x007FFFFFu; den[i] = 0.0f; }   // fkey(-inf)
}

// ---------- pass 1: segment max of leaky_relu(alpha_s[src]+alpha_d[dst]) ----------
__global__ void edge_max(const int* __restrict__ src, const int* __restrict__ dst,
                         const float* __restrict__ als, const float* __restrict__ ald,
                         unsigned* __restrict__ ek, int E)
{
    int i = blockIdx.x * blockDim.x + threadIdx.x;
    if (i >= E) return;
    float ev = lrelu(als[src[i]] + ald[dst[i]]);
    atomicMax(&ek[dst[i]], fkey(ev));
}

// ---------- emax fixup: decode key, -inf (no incoming edges) -> 0 ----------
__global__ void fixup_emax(unsigned* __restrict__ ek, int nd)
{
    int i = blockIdx.x * blockDim.x + threadIdx.x;
    if (i >= nd) return;
    float m = keyf(ek[i]);
    if (m < -3.0e38f) m = 0.0f;
    ((float*)ek)[i] = m;
}

// ---------- pass 2: one wave per edge; w=exp(e-emax); scatter w*hs[src] ----------
__global__ void edge_msg(const int* __restrict__ src, const int* __restrict__ dst,
                         const float* __restrict__ als, const float* __restrict__ ald,
                         const float* __restrict__ emax, const float* __restrict__ hsrc,
                         float* __restrict__ acc, float* __restrict__ den, int E)
{
    int e    = (int)((blockIdx.x * blockDim.x + threadIdx.x) >> 5);
    int lane = (int)(threadIdx.x & 31);
    if (e >= E) return;
    int s = src[e], d = dst[e];
    float ev  = lrelu(als[s] + ald[d]);
    float wgt = __expf(ev - emax[d]);
    if (lane == 0) atomicAdd(&den[d], wgt);
    float4 h4 = ((const float4*)(hsrc + (size_t)s * D))[lane];
    float* arow = acc + (size_t)d * D + lane * 4;
    atomicAdd(arow + 0, wgt * h4.x);
    atomicAdd(arow + 1, wgt * h4.y);
    atomicAdd(arow + 2, wgt * h4.z);
    atomicAdd(arow + 3, wgt * h4.w);
}

// ---------- finalize: msg/den + bias, with conv-specific combine/ReLU ----------
__global__ void finalize(float* __restrict__ outrow, const float* __restrict__ acc,
                         const float* __restrict__ den, const float* __restrict__ bvec,
                         int nd, int mode)
{
    long i = (long)blockIdx.x * blockDim.x + threadIdx.x;
    if (i >= (long)nd * D) return;
    int n = (int)(i >> 7), h = (int)(i & (D - 1));
    float v = acc[i] / fmaxf(den[n], 1e-16f) + bvec[h];
    if (mode == 0)      outrow[i] = v;                          // first artwork conv
    else if (mode == 1) outrow[i] = fmaxf(outrow[i] + v, 0.f);  // HeteroConv sum + ReLU
    else                outrow[i] = fmaxf(v, 0.f);              // artist conv + ReLU
}

static inline unsigned gdim(long n, int b) { return (unsigned)((n + b - 1) / b); }

extern "C" void kernel_launch(void* const* d_in, const int* in_sizes, int n_in,
                              void* d_out, int out_size, void* d_ws, size_t ws_size,
                              hipStream_t stream)
{
    const float* xa   = (const float*)d_in[0];
    const float* xw   = (const float*)d_in[1];
    const int* src_ca = (const int*)d_in[2];
    const int* dst_ca = (const int*)d_in[3];
    const int* src_ac = (const int*)d_in[4];
    const int* dst_ac = (const int*)d_in[5];
    const int* src_ww = (const int*)d_in[6];
    const int* dst_ww = (const int*)d_in[7];
    const float* Ws   = (const float*)d_in[8];
    const float* Wd   = (const float*)d_in[9];
    const float* a_s  = (const float*)d_in[10];
    const float* a_d  = (const float*)d_in[11];
    const float* bias = (const float*)d_in[12];
    float* out = (float*)d_out;

    const int Na  = in_sizes[0] / D;
    const int Nw  = in_sizes[1] / D;
    const int Eca = in_sizes[2], Eac = in_sizes[4], Eww = in_sizes[6];
    const long Nmax = (Na > Nw) ? Na : Nw;
    (void)n_in; (void)out_size; (void)ws_size;

    // ---- carve workspace (all L2-resident hot structures) ----
    char* wbase = (char*)d_ws;
    size_t off = 0;
    auto carve = [&](size_t bytes) -> void* {
        void* p = wbase + off;
        off = (off + bytes + 255) & ~(size_t)255;
        return p;
    };
    unsigned short* xa_bf = (unsigned short*)carve((size_t)Na * D * 2);
    unsigned short* xw_bf = (unsigned short*)carve((size_t)Nw * D * 2);
    unsigned short* wsw   = (unsigned short*)carve((size_t)3 * D * D * 2);
    float* us   = (float*)carve((size_t)3 * D * 4);
    float* ud   = (float*)carve((size_t)3 * D * 4);
    float* hs   = (float*)carve((size_t)Nmax * D * 4);   // reused across convs
    float* accw = (float*)carve((size_t)Nw * D * 4);     // conv-ww accumulator
    float* als  = (float*)carve((size_t)Nmax * 4);
    float* ald  = (float*)carve((size_t)Nmax * 4);
    unsigned* ek = (unsigned*)carve((size_t)Nmax * 4);   // emax keys, then floats
    float* den  = (float*)carve((size_t)Nmax * 4);

    // ---- prep ----
    prep_u   <<<gdim(6 * D, 256),     256, 0, stream>>>(Ws, Wd, a_s, a_d, us, ud);
    swizzle_w<<<gdim(3 * D * D, 256), 256, 0, stream>>>(Ws, wsw);
    cvt_bf16 <<<gdim((long)Na * D, 256), 256, 0, stream>>>(xa, xa_bf, (long)Na * D);
    cvt_bf16 <<<gdim((long)Nw * D, 256), 256, 0, stream>>>(xw, xw_bf, (long)Nw * D);

    auto conv = [&](const float* xs, const unsigned short* xsb, int Ns,
                    const float* xd, int Nd,
                    const int* src, const int* dst, int E, int t,
                    float* acc, float* outrow, int mode) {
        long waves = (Ns + 15) / 16;
        gat_gemm_wmma<<<gdim(waves * 32, 256), 256, 0, stream>>>(xsb, wsw + (size_t)t * D * D, hs, Ns);
        matvec<<<gdim((long)Ns * 32, 256), 256, 0, stream>>>(xs, us + t * D, als, Ns);
        matvec<<<gdim((long)Nd * 32, 256), 256, 0, stream>>>(xd, ud + t * D, ald, Nd);
        reset_conv<<<gdim((long)Nd * D, 256), 256, 0, stream>>>(ek, den, acc, Nd);
        edge_max<<<gdim(E, 256), 256, 0, stream>>>(src, dst, als, ald, ek, E);
        fixup_emax<<<gdim(Nd, 256), 256, 0, stream>>>(ek, Nd);
        edge_msg<<<gdim((long)E * 32, 256), 256, 0, stream>>>(src, dst, als, ald,
                                                              (const float*)ek, hs, acc, den, E);
        finalize<<<gdim((long)Nd * D, 256), 256, 0, stream>>>(outrow, acc, den, bias + t * D, Nd, mode);
    };

    float* out_aw = out + (size_t)Na * D;
    // edge type 0: artist -> artwork (accumulate directly in d_out artwork rows)
    conv(xa, xa_bf, Na, xw, Nw, src_ca, dst_ca, Eca, 0, out_aw, out_aw, 0);
    // edge type 2: artwork -> artwork (HeteroConv sum into artwork rows, then ReLU)
    conv(xw, xw_bf, Nw, xw, Nw, src_ww, dst_ww, Eww, 2, accw, out_aw, 1);
    // edge type 1: artwork -> artist (accumulate in d_out artist rows, ReLU)
    conv(xw, xw_bf, Nw, xa, Na, src_ac, dst_ac, Eac, 1, out, out, 2);
}
